// SigLipLoss_1116691497359
// MI455X (gfx1250) — compile-verified, matrix operand order checked by reading, not run on hardware
//
#include <hip/hip_runtime.h>
#include <hip/hip_bf16.h>
#include <math.h>

typedef __attribute__((ext_vector_type(16))) __bf16 v16bf;
typedef __attribute__((ext_vector_type(8)))  float  v8f;
typedef __attribute__((ext_vector_type(4)))  int    v4i;

#define TILE      128   // block output tile edge (M and N)
#define DDIM      256   // feature dim (K)
#define KCHUNK     64   // K staged per LDS round
#define NCHUNK    (DDIM / KCHUNK)
#define LDS_PITCH  72   // 64 + 8 ushort pad (breaks 64-bank row alignment)

// Direct global -> LDS async DMA (gfx1250), tracked by ASYNCcnt.
#if defined(__has_builtin)
#if __has_builtin(__builtin_amdgcn_global_load_async_to_lds_b128) && \
    __has_builtin(__builtin_amdgcn_s_wait_asynccnt)
#define USE_ASYNC_LDS 1
#endif
#endif
#ifndef USE_ASYNC_LDS
#define USE_ASYNC_LDS 0
#endif

#if USE_ASYNC_LDS
__device__ __forceinline__ void async_b128(const void* g, void* l) {
    __builtin_amdgcn_global_load_async_to_lds_b128(
        (__attribute__((address_space(1))) v4i*)g,
        (__attribute__((address_space(3))) v4i*)l, 0, 0);
}
#endif

// fp32 pair -> packed bf16x2: round-half-up then single v_perm_b32 byte gather
__device__ __forceinline__ unsigned int pack_bf16x2(float lo, float hi) {
    unsigned int ul = __float_as_uint(lo) + 0x8000u;
    unsigned int uh = __float_as_uint(hi) + 0x8000u;
    return __builtin_amdgcn_perm(uh, ul, 0x07060302u);  // {uh[31:16], ul[31:16]}
}

union FragU { v16bf v; uint4 q[2]; };

// ---------------------------------------------------------------------------
// Kernel 1: one-shot fp32 -> bf16 conversion of both matrices into d_ws.
// ---------------------------------------------------------------------------
__global__ __launch_bounds__(256) void convert_bf16(
    const float* __restrict__ img, const float* __restrict__ txt,
    unsigned short* __restrict__ imgb, unsigned short* __restrict__ txtb) {
    const size_t i = (size_t)blockIdx.x * blockDim.x + threadIdx.x;  // float4 index
    float4 a = ((const float4*)img)[i];
    float4 b = ((const float4*)txt)[i];
    ((uint2*)imgb)[i] = make_uint2(pack_bf16x2(a.x, a.y), pack_bf16x2(a.z, a.w));
    ((uint2*)txtb)[i] = make_uint2(pack_bf16x2(b.x, b.y), pack_bf16x2(b.z, b.w));
}

// ---------------------------------------------------------------------------
// Kernel 2: diagonal correction. log_sig(z) - log_sig(-z) == z, so the
// diagonal contribution is scale*trace(img@txt^T) + N*bias.
// ---------------------------------------------------------------------------
__global__ __launch_bounds__(256) void trace_diag(
    const float* __restrict__ img, const float* __restrict__ txt,
    const float* __restrict__ scale_p, const float* __restrict__ bias_p,
    float* __restrict__ out, float invN) {
    const size_t i = (size_t)blockIdx.x * blockDim.x + threadIdx.x;  // float4 index
    float4 a = ((const float4*)img)[i];
    float4 b = ((const float4*)txt)[i];
    float s = a.x * b.x + a.y * b.y + a.z * b.z + a.w * b.w;
#pragma unroll
    for (int off = 16; off > 0; off >>= 1) s += __shfl_xor(s, off, 32);
    if ((threadIdx.x & 31) == 0)
        unsafeAtomicAdd(out, -s * (*scale_p) * invN);
    if (blockIdx.x == 0 && threadIdx.x == 0)
        unsafeAtomicAdd(out, -(*bias_p));   // the N*bias/N term, once
}

// ---------------------------------------------------------------------------
// Staging helpers (register path)
// ---------------------------------------------------------------------------
__device__ __forceinline__ void load_regs_bf16(const unsigned short* __restrict__ g,
                                               int gRow0, int kc, int t, uint4 (&q)[4]) {
    const int row = t >> 1, seg = t & 1;
    const uint4* src =
        (const uint4*)(g + (size_t)(gRow0 + row) * DDIM + kc * KCHUNK + seg * 32);
#pragma unroll
    for (int i = 0; i < 4; ++i) q[i] = src[i];
}
__device__ __forceinline__ void load_regs_f32(const float* __restrict__ g,
                                              int gRow0, int kc, int t, uint4 (&q)[4]) {
    const int row = t >> 1, seg = t & 1;
    const float4* src =
        (const float4*)(g + (size_t)(gRow0 + row) * DDIM + kc * KCHUNK + seg * 32);
    float4 f[8];
#pragma unroll
    for (int i = 0; i < 8; ++i) f[i] = src[i];
#pragma unroll
    for (int i = 0; i < 4; ++i)
        q[i] = make_uint4(pack_bf16x2(f[2 * i].x, f[2 * i].y),
                          pack_bf16x2(f[2 * i].z, f[2 * i].w),
                          pack_bf16x2(f[2 * i + 1].x, f[2 * i + 1].y),
                          pack_bf16x2(f[2 * i + 1].z, f[2 * i + 1].w));
}
__device__ __forceinline__ void store_chunk_lds(const uint4 (&q)[4],
                                                unsigned short (*lds)[LDS_PITCH], int t) {
    const int row = t >> 1, seg = t & 1;
    uint4* dst = (uint4*)&lds[row][seg * 32];
#pragma unroll
    for (int i = 0; i < 4; ++i) dst[i] = q[i];
}

#if USE_ASYNC_LDS
// Async DMA staging: each thread copies 64B (4 x b128) of a TILE x KCHUNK chunk.
__device__ __forceinline__ void issue_chunk_async(const unsigned short* __restrict__ g,
                                                  int gRow0, int kc,
                                                  unsigned short (*lds)[LDS_PITCH], int t) {
    const int row = t >> 1, seg = t & 1;
    const unsigned short* src = g + (size_t)(gRow0 + row) * DDIM + kc * KCHUNK + seg * 32;
    unsigned short* dst = &lds[row][seg * 32];
#pragma unroll
    for (int i = 0; i < 4; ++i)
        async_b128(src + i * 8, dst + i * 8);
}
#endif

// One KCHUNK of WMMA work for a wave's 32x64 tile.
__device__ __forceinline__ void mma_chunk(const unsigned short (*A)[LDS_PITCH],
                                          const unsigned short (*B)[LDS_PITCH],
                                          int waveM, int waveN, int half, int r16,
                                          v8f (&acc)[2][4]) {
#pragma unroll
    for (int kk = 0; kk < KCHUNK; kk += 32) {
        const int k0 = kk + half * 8;
        FragU afr[2], bfr[4];
#pragma unroll
        for (int i = 0; i < 2; ++i) {
            const unsigned short* p = &A[waveM * 32 + i * 16 + r16][k0];
            afr[i].q[0] = *(const uint4*)p;
            afr[i].q[1] = *(const uint4*)(p + 16);
        }
#pragma unroll
        for (int j = 0; j < 4; ++j) {
            const unsigned short* p = &B[waveN * 64 + j * 16 + r16][k0];
            bfr[j].q[0] = *(const uint4*)p;
            bfr[j].q[1] = *(const uint4*)(p + 16);
        }
#pragma unroll
        for (int i = 0; i < 2; ++i)
#pragma unroll
            for (int j = 0; j < 4; ++j)
                acc[i][j] = __builtin_amdgcn_wmma_f32_16x16x32_bf16(
                    false, afr[i].v, false, bfr[j].v,
                    (short)0, acc[i][j], false, false);
    }
}

// ---------------------------------------------------------------------------
// Kernel 3: fused GEMM + uniform-label log-sigmoid reduction.
// ---------------------------------------------------------------------------
template <bool PRE>
__global__ __launch_bounds__(256) void siglip_gemm_loss(
    const void* __restrict__ aRaw, const void* __restrict__ bRaw,
    const float* __restrict__ scale_p, const float* __restrict__ bias_p,
    float* __restrict__ out, float invN) {
    __shared__ unsigned short As[2][TILE][LDS_PITCH];   // 2 x 18KB
    __shared__ unsigned short Bs[2][TILE][LDS_PITCH];   // 2 x 18KB

    const int t     = threadIdx.x;
    const int lane  = t & 31;
    const int wave  = t >> 5;       // 8 wave32s
    const int waveM = wave & 3;     // 4 x 32-row strips
    const int waveN = wave >> 2;    // 2 x 64-col strips  -> 32x64 per wave
    const int half  = lane >> 4;
    const int r16   = lane & 15;

    const int tileM = blockIdx.x * TILE;
    const int tileN = blockIdx.y * TILE;

    v8f acc[2][4] = {};

#if USE_ASYNC_LDS
    if constexpr (PRE) {
        // Async double-buffer: DMA chunk kc+1 global->LDS while WMMAs chew kc.
        const unsigned short* aBf = (const unsigned short*)aRaw;
        const unsigned short* bBf = (const unsigned short*)bRaw;
        issue_chunk_async(aBf, tileM, 0, As[0], t);
        issue_chunk_async(bBf, tileN, 0, Bs[0], t);
#pragma unroll
        for (int kc = 0; kc < NCHUNK; ++kc) {
            __builtin_amdgcn_s_wait_asynccnt(0);   // my DMA into buf[kc&1] done
            __syncthreads();                        // everyone's DMA visible
            if (kc + 1 < NCHUNK) {
                issue_chunk_async(aBf, tileM, kc + 1, As[(kc + 1) & 1], t);
                issue_chunk_async(bBf, tileN, kc + 1, Bs[(kc + 1) & 1], t);
            }
            mma_chunk(As[kc & 1], Bs[kc & 1], waveM, waveN, half, r16, acc);
        }
    } else
#endif
    {
        uint4 qa[4], qb[4];
        if (PRE) {
            load_regs_bf16((const unsigned short*)aRaw, tileM, 0, t, qa);
            load_regs_bf16((const unsigned short*)bRaw, tileN, 0, t, qb);
        } else {
            load_regs_f32((const float*)aRaw, tileM, 0, t, qa);
            load_regs_f32((const float*)bRaw, tileN, 0, t, qb);
        }
        store_chunk_lds(qa, As[0], t);
        store_chunk_lds(qb, Bs[0], t);
#pragma unroll
        for (int kc = 0; kc < NCHUNK; ++kc) {
            __syncthreads();
            if (kc + 1 < NCHUNK) {
                if (PRE) {
                    load_regs_bf16((const unsigned short*)aRaw, tileM, kc + 1, t, qa);
                    load_regs_bf16((const unsigned short*)bRaw, tileN, kc + 1, t, qb);
                } else {
                    load_regs_f32((const float*)aRaw, tileM, kc + 1, t, qa);
                    load_regs_f32((const float*)bRaw, tileN, kc + 1, t, qb);
                }
            }
            mma_chunk(As[kc & 1], Bs[kc & 1], waveM, waveN, half, r16, acc);
            if (kc + 1 < NCHUNK) {
                store_chunk_lds(qa, As[(kc + 1) & 1], t);
                store_chunk_lds(qb, Bs[(kc + 1) & 1], t);
            }
        }
    }

    // Uniform label = -1 (diagonal corrected by trace_diag):
    // x = -z, log_sigmoid(x) = min(x,0) - log(1 + exp(-|x|)).
    const float scale = *scale_p;
    const float bias  = *bias_p;
    float partial = 0.f;
#pragma unroll
    for (int i = 0; i < 2; ++i)
#pragma unroll
        for (int j = 0; j < 4; ++j)
#pragma unroll
            for (int rr = 0; rr < 8; ++rr) {
                const float x = -fmaf(scale, acc[i][j][rr], bias);
                partial += fminf(x, 0.f) - __logf(1.f + __expf(-fabsf(x)));
            }
#pragma unroll
    for (int off = 16; off > 0; off >>= 1)
        partial += __shfl_xor(partial, off, 32);
    if (lane == 0)
        unsafeAtomicAdd(out, -partial * invN);
}

extern "C" void kernel_launch(void* const* d_in, const int* in_sizes, int n_in,
                              void* d_out, int out_size, void* d_ws, size_t ws_size,
                              hipStream_t stream) {
    (void)n_in; (void)out_size;
    const float* img     = (const float*)d_in[0];
    const float* txt     = (const float*)d_in[1];
    const float* scale_p = (const float*)d_in[2];
    const float* bias_p  = (const float*)d_in[3];
    float* out = (float*)d_out;

    const int    n     = in_sizes[0] / DDIM;    // 16384
    const float  invN  = 1.0f / (float)n;
    const size_t elems = (size_t)n * DDIM;

    (void)hipMemsetAsync(out, 0, sizeof(float), stream);

    const int vecBlocks = (int)(elems / 4 / 256);
    trace_diag<<<vecBlocks, 256, 0, stream>>>(img, txt, scale_p, bias_p, out, invN);

    dim3 grid(n / TILE, n / TILE);
    const size_t need = 2 * elems * sizeof(unsigned short);   // 16 MB of bf16
    if (ws_size >= need) {
        unsigned short* imgb = (unsigned short*)d_ws;
        unsigned short* txtb = imgb + elems;
        convert_bf16<<<vecBlocks, 256, 0, stream>>>(img, txt, imgb, txtb);
        siglip_gemm_loss<true><<<grid, 256, 0, stream>>>(imgb, txtb, scale_p, bias_p, out, invN);
    } else {
        siglip_gemm_loss<false><<<grid, 256, 0, stream>>>(img, txt, scale_p, bias_p, out, invN);
    }
}
